// Translator_28613072126537
// MI455X (gfx1250) — matrix operand order repaired and checked by reading, not compiled, not run on hardware
//
#include <hip/hip_runtime.h>
#include <math.h>

typedef __attribute__((ext_vector_type(16))) __bf16 v16bf;
typedef __attribute__((ext_vector_type(8)))  float  v8f;

#define DG   256
#define DG3  16777216
#define NCH  8

__device__ __forceinline__ int clampi(int v, int lo, int hi) {
  return v < lo ? lo : (v > hi ? hi : v);
}
__device__ __forceinline__ unsigned encf(float f) {
  unsigned u = __float_as_uint(f);
  return (u & 0x80000000u) ? ~u : (u | 0x80000000u);
}
__device__ __forceinline__ float decf(unsigned e) {
  unsigned u = (e & 0x80000000u) ? (e & 0x7fffffffu) : ~e;
  return __uint_as_float(u);
}
__device__ __forceinline__ __bf16 f2bf(float f) { return (__bf16)f; }

// ---------------- Pass 1: global max of gathered channel-7 values ----------
__global__ __launch_bounds__(256)
void kmax_kernel(const float* __restrict__ points, const float* __restrict__ grid,
                 unsigned* __restrict__ slot, int n) {
  int p = blockIdx.x * 256 + threadIdx.x;
  int pc = p < n ? p : n - 1;
  int ix = clampi((int)floorf(points[3 * pc + 0]), 0, DG - 1);
  int iy = clampi((int)floorf(points[3 * pc + 1]), 0, DG - 1);
  int iz = clampi((int)floorf(points[3 * pc + 2]), 0, DG - 1);
  int xs[3], ys[3], zs[3];
  xs[0] = clampi(ix - 1, 0, DG - 1) << 16; xs[1] = ix << 16; xs[2] = clampi(ix + 1, 0, DG - 1) << 16;
  ys[0] = clampi(iy - 1, 0, DG - 1) << 8;  ys[1] = iy << 8;  ys[2] = clampi(iy + 1, 0, DG - 1) << 8;
  zs[0] = clampi(iz - 1, 0, DG - 1);       zs[1] = iz;       zs[2] = clampi(iz + 1, 0, DG - 1);
  const float* g7 = grid + (size_t)7 * DG3;   // uniform base -> saddr form
  float mx = -3.4e38f;
  #pragma unroll
  for (int k = 0; k < 27; ++k) {
    int b = xs[k / 9] + ys[(k / 3) % 3] + zs[k % 3];
    mx = fmaxf(mx, g7[b]);
  }
  __shared__ float red[256];
  red[threadIdx.x] = mx;
  __syncthreads();
  for (int s = 128; s > 0; s >>= 1) {
    if (threadIdx.x < s) red[threadIdx.x] = fmaxf(red[threadIdx.x], red[threadIdx.x + s]);
    __syncthreads();
  }
  if (threadIdx.x == 0) atomicMax(slot, encf(red[0]));
}

// --------------- dense layer on concat(a[IN1], b[IN2]) ---------------------
template <int IN1, int IN2, int OUT>
__device__ __forceinline__ void dense2(const float* __restrict__ a,
                                       const float* __restrict__ b,
                                       const float* __restrict__ wlds,
                                       const float* __restrict__ blds,
                                       float* __restrict__ out) {
  #pragma unroll
  for (int o = 0; o < OUT; ++o) out[o] = blds[o];
  #pragma unroll
  for (int i = 0; i < IN1; ++i) {
    float x = a[i];
    const float4* w4 = (const float4*)(wlds + i * OUT);
    #pragma unroll
    for (int q = 0; q < OUT / 4; ++q) {
      float4 w = w4[q];
      out[4 * q + 0] += x * w.x; out[4 * q + 1] += x * w.y;
      out[4 * q + 2] += x * w.z; out[4 * q + 3] += x * w.w;
    }
  }
  #pragma unroll
  for (int i = 0; i < IN2; ++i) {
    float x = b[i];
    const float4* w4 = (const float4*)(wlds + (IN1 + i) * OUT);
    #pragma unroll
    for (int q = 0; q < OUT / 4; ++q) {
      float4 w = w4[q];
      out[4 * q + 0] += x * w.x; out[4 * q + 1] += x * w.y;
      out[4 * q + 2] += x * w.z; out[4 * q + 3] += x * w.w;
    }
  }
  #pragma unroll
  for (int o = 0; o < OUT; ++o) out[o] = tanhf(out[o]);
}

// ---------------- Pass 2: fused gather + WMMA ctx GEMM + MLP ---------------
__global__ __launch_bounds__(256)
void fused_kernel(const float* __restrict__ points, const float* __restrict__ grid,
                  const float* __restrict__ Wc, const float* __restrict__ bc,
                  const float* __restrict__ W1, const float* __restrict__ b1,
                  const float* __restrict__ W2, const float* __restrict__ b2,
                  const float* __restrict__ W3, const float* __restrict__ b3,
                  const float* __restrict__ W4, const float* __restrict__ b4,
                  const float* __restrict__ Wsd, const float* __restrict__ bsd,
                  const float* __restrict__ Woc, const float* __restrict__ boc,
                  const unsigned* __restrict__ slot,
                  float* __restrict__ out, int n) {
  __shared__ __align__(32) __bf16 sB[7][32][16];        // B fragments (Wc, row-7 prescaled)
  __shared__ __align__(16) float sCtx[256][8];
  __shared__ __align__(16) float sW1[512];
  __shared__ __align__(16) float sW2[640];
  __shared__ __align__(16) float sW3[192];
  __shared__ __align__(16) float sW4[128];
  __shared__ float sb1[32], sb2[16], sb3[8], sb4[8], sWs[8], sWo[8], sBc[8];

  const int tid = threadIdx.x;
  const float gmax = decf(*slot);
  const float inv_max = 1.0f / (gmax + 1e-9f);

  for (int i = tid; i < 512; i += 256) sW1[i] = W1[i];
  for (int i = tid; i < 640; i += 256) sW2[i] = W2[i];
  for (int i = tid; i < 192; i += 256) sW3[i] = W3[i];
  for (int i = tid; i < 128; i += 256) sW4[i] = W4[i];
  if (tid < 32) sb1[tid] = b1[tid];
  if (tid < 16) sb2[tid] = b2[tid];
  if (tid < 8) {
    sb3[tid] = b3[tid]; sb4[tid] = b4[tid];
    sWs[tid] = Wsd[tid]; sWo[tid] = Woc[tid]; sBc[tid] = bc[tid];
  }
  // B layout: lanes 0-15 hold K = 32*ch + {0..7, 16..23}, lanes 16-31 hold K = 32*ch + {8..15, 24..31},
  // column N = lane & 15 (cols 8..15 zero-padded; K >= 216 zero-padded).
  // Channel-7 rows (K%8==7) are pre-scaled by 1/max_count -> gather needs no scaling,
  // and the K>=216 zero rows absorb any garbage A tail values.
  for (int i = tid; i < 7 * 32 * 16; i += 256) {
    int ch = i >> 9;
    int rem = i & 511;
    int ln = rem >> 4;
    int j = rem & 15;
    int ncol = ln & 15;
    int K = 32 * ch + ((j >> 3) << 4) + ((ln >> 4) << 3) + (j & 7);
    float v = (ncol < 8 && K < 216) ? Wc[K * 8 + ncol] : 0.0f;
    if ((K & 7) == 7) v *= inv_max;
    sB[ch][ln][j] = f2bf(v);
  }
  __syncthreads();

  const int lane = tid & 31;
  const int wave = tid >> 5;
  const bool hi = lane >= 16;
  const int m = lane & 15;
  const int blockBase = blockIdx.x * 256;

  // Uniform channel-slab bases (lower to saddr-form global loads).
  const float* gch[8];
  #pragma unroll
  for (int c = 0; c < 8; ++c) gch[c] = grid + (size_t)c * DG3;

  // Coordinates of this lane's point in each of the two 16-point tiles.
  int p0 = blockBase + wave * 32 + m;       if (p0 > n - 1) p0 = n - 1;
  int p1 = blockBase + wave * 32 + 16 + m;  if (p1 > n - 1) p1 = n - 1;
  int ix0 = clampi((int)floorf(points[3 * p0 + 0]), 0, DG - 1);
  int iy0 = clampi((int)floorf(points[3 * p0 + 1]), 0, DG - 1);
  int iz0 = clampi((int)floorf(points[3 * p0 + 2]), 0, DG - 1);
  int ix1 = clampi((int)floorf(points[3 * p1 + 0]), 0, DG - 1);
  int iy1 = clampi((int)floorf(points[3 * p1 + 1]), 0, DG - 1);
  int iz1 = clampi((int)floorf(points[3 * p1 + 2]), 0, DG - 1);

  // Pre-shifted, clamped coordinate components (constant-indexed after unroll).
  int xs0[3], ys0[3], zs0[3], xs1[3], ys1[3], zs1[3];
  xs0[0] = clampi(ix0 - 1, 0, DG - 1) << 16; xs0[1] = ix0 << 16; xs0[2] = clampi(ix0 + 1, 0, DG - 1) << 16;
  ys0[0] = clampi(iy0 - 1, 0, DG - 1) << 8;  ys0[1] = iy0 << 8;  ys0[2] = clampi(iy0 + 1, 0, DG - 1) << 8;
  zs0[0] = clampi(iz0 - 1, 0, DG - 1);       zs0[1] = iz0;       zs0[2] = clampi(iz0 + 1, 0, DG - 1);
  xs1[0] = clampi(ix1 - 1, 0, DG - 1) << 16; xs1[1] = ix1 << 16; xs1[2] = clampi(ix1 + 1, 0, DG - 1) << 16;
  ys1[0] = clampi(iy1 - 1, 0, DG - 1) << 8;  ys1[1] = iy1 << 8;  ys1[2] = clampi(iy1 + 1, 0, DG - 1) << 8;
  zs1[0] = clampi(iz1 - 1, 0, DG - 1);       zs1[1] = iz1;       zs1[2] = clampi(iz1 + 1, 0, DG - 1);

  v8f acc0 = {0.f, 0.f, 0.f, 0.f, 0.f, 0.f, 0.f, 0.f};
  v8f acc1 = {0.f, 0.f, 0.f, 0.f, 0.f, 0.f, 0.f, 0.f};

  // K chunks of 32 = 4 neighbors each; lane m handles even neighbors of point m,
  // lane m+16 handles odd neighbors (matches the 16-bit A-fragment layout).
  #pragma unroll
  for (int ch = 0; ch < 7; ++ch) {
    v16bf b = *(const v16bf*)(&sB[ch][lane][0]);
    v16bf a0, a1;
    #pragma unroll
    for (int h = 0; h < 2; ++h) {
      const int ke = 4 * ch + 2 * h;                 // even-lane neighbor (compile-time)
      const int ko = (ke + 1 > 26) ? 26 : ke + 1;    // odd-lane neighbor (k=27 -> dup of 26; B row is 0)
      int be0 = xs0[ke / 9] + ys0[(ke / 3) % 3] + zs0[ke % 3];
      int bo0 = xs0[ko / 9] + ys0[(ko / 3) % 3] + zs0[ko % 3];
      int bb0 = hi ? bo0 : be0;
      int be1 = xs1[ke / 9] + ys1[(ke / 3) % 3] + zs1[ke % 3];
      int bo1 = xs1[ko / 9] + ys1[(ko / 3) % 3] + zs1[ko % 3];
      int bb1 = hi ? bo1 : be1;
      #pragma unroll
      for (int c = 0; c < NCH; ++c) {
        a0[8 * h + c] = f2bf(gch[c][bb0]);
        a1[8 * h + c] = f2bf(gch[c][bb1]);
      }
    }
    acc0 = __builtin_amdgcn_wmma_f32_16x16x32_bf16(false, a0, false, b,
                                                   (short)0, acc0, false, false);
    acc1 = __builtin_amdgcn_wmma_f32_16x16x32_bf16(false, a1, false, b,
                                                   (short)0, acc1, false, false);
  }

  // Bias + LayerNorm + tanh in the C-matrix layout (channels across 8 lanes).
  float bias = (m < 8) ? sBc[m] : 0.0f;
  #pragma unroll
  for (int r = 0; r < 8; ++r) {
    float z0 = acc0[r] + bias;
    float z1 = acc1[r] + bias;
    float s0 = z0, s1 = z1;
    s0 += __shfl_xor(s0, 1, 32); s0 += __shfl_xor(s0, 2, 32); s0 += __shfl_xor(s0, 4, 32);
    s1 += __shfl_xor(s1, 1, 32); s1 += __shfl_xor(s1, 2, 32); s1 += __shfl_xor(s1, 4, 32);
    float d0 = z0 - s0 * 0.125f;
    float d1 = z1 - s1 * 0.125f;
    float v0 = d0 * d0, v1 = d1 * d1;
    v0 += __shfl_xor(v0, 1, 32); v0 += __shfl_xor(v0, 2, 32); v0 += __shfl_xor(v0, 4, 32);
    v1 += __shfl_xor(v1, 1, 32); v1 += __shfl_xor(v1, 2, 32); v1 += __shfl_xor(v1, 4, 32);
    float c0v = tanhf(d0 * rsqrtf(v0 * 0.125f + 1e-5f));
    float c1v = tanhf(d1 * rsqrtf(v1 * 0.125f + 1e-5f));
    if (m < 8) {
      int row0 = wave * 32 + (hi ? 8 + r : r);
      sCtx[row0][m] = c0v;
      sCtx[row0 + 16][m] = c1v;
    }
  }
  __syncthreads();

  // Stage 2: one point per lane.
  int q = blockBase + tid;
  int qc = q > n - 1 ? n - 1 : q;
  int cx = clampi((int)floorf(points[3 * qc + 0]), 0, DG - 1);
  int cy = clampi((int)floorf(points[3 * qc + 1]), 0, DG - 1);
  int cz = clampi((int)floorf(points[3 * qc + 2]), 0, DG - 1);
  int cb = (cx << 16) + (cy << 8) + cz;
  float cen[8];
  #pragma unroll
  for (int c = 0; c < 8; ++c) cen[c] = gch[c][cb];
  cen[7] *= inv_max;
  float ctx[8];
  #pragma unroll
  for (int c = 0; c < 8; ++c) ctx[c] = sCtx[tid][c];

  float h1[32]; dense2<8, 8, 32>(cen, ctx, sW1, sb1, h1);
  float h2[16]; dense2<8, 32, 16>(cen, h1, sW2, sb2, h2);
  float h3[8];  dense2<8, 16, 8>(cen, h2, sW3, sb3, h3);
  float h4[8];  dense2<8, 8, 8>(cen, h3, sW4, sb4, h4);

  float ssdf = bsd[0], socc = boc[0];
  #pragma unroll
  for (int c = 0; c < 8; ++c) { ssdf += h4[c] * sWs[c]; socc += h4[c] * sWo[c]; }
  float sdf = tanhf(ssdf);
  float occ = 1.0f / (1.0f + expf(-socc));
  if (q < n) {
    out[2 * q + 0] = sdf;
    out[2 * q + 1] = occ;
  }
}

extern "C" void kernel_launch(void* const* d_in, const int* in_sizes, int n_in,
                              void* d_out, int out_size, void* d_ws, size_t ws_size,
                              hipStream_t stream) {
  const float* points = (const float*)d_in[0];
  const float* grid   = (const float*)d_in[1];
  const float* Wc = (const float*)d_in[2];
  const float* bc = (const float*)d_in[3];
  const float* W1 = (const float*)d_in[4];
  const float* b1 = (const float*)d_in[5];
  const float* W2 = (const float*)d_in[6];
  const float* b2 = (const float*)d_in[7];
  const float* W3 = (const float*)d_in[8];
  const float* b3 = (const float*)d_in[9];
  const float* W4 = (const float*)d_in[10];
  const float* b4 = (const float*)d_in[11];
  const float* Ws = (const float*)d_in[12];
  const float* bs = (const float*)d_in[13];
  const float* Wo = (const float*)d_in[14];
  const float* bo = (const float*)d_in[15];
  (void)n_in; (void)out_size; (void)ws_size;

  int n = in_sizes[0] / 3;
  unsigned* slot = (unsigned*)d_ws;
  hipMemsetAsync(d_ws, 0, sizeof(unsigned), stream);

  int nb = (n + 255) / 256;
  kmax_kernel<<<nb, 256, 0, stream>>>(points, grid, slot, n);
  fused_kernel<<<nb, 256, 0, stream>>>(points, grid, Wc, bc, W1, b1, W2, b2,
                                       W3, b3, W4, b4, Ws, bs, Wo, bo,
                                       slot, (float*)d_out, n);
}
